// DeepseekV2MoE_47802986004843
// MI455X (gfx1250) — matrix operand order repaired and checked by reading, not compiled, not run on hardware
//
#include <hip/hip_runtime.h>
#include <hip/hip_bf16.h>
#include <stdint.h>

// ---------------------------------------------------------------------------
// DeepseekV2 MoE for MI455X (gfx1250): bf16 WMMA GEMMs, double-buffered async
// A-tile staging (ASYNCcnt), packed bf16 B-tile conversion, fused SiLU.
// ---------------------------------------------------------------------------

typedef __bf16 bf16;
typedef __attribute__((ext_vector_type(16))) __bf16 v16bf;
typedef __attribute__((ext_vector_type(8)))  __bf16 bf16x8;
typedef __attribute__((ext_vector_type(8)))  float  v8f;
typedef __attribute__((ext_vector_type(4)))  float  f32x4;
typedef __attribute__((ext_vector_type(4)))  uint32_t u32x4;

#define T_  2048
#define H_  2048
#define I_  1408
#define E_  16
#define K_  6
#define G_  4
#define IS_ 2816

#define BM  128   // workgroup M tile
#define BN  32    // workgroup N tile (per matrix)
#define BK  32    // K step (matches 16x16x32 bf16 WMMA)
#define LDP 40    // padded LDS row length (elems) -> 80B stride, 16B aligned

#define ABUF  (BM * LDP)          // A buffer stride, bf16 elems
#define BBUF  ((BN * LDP) / 2)    // B buffer stride, uint32 elems

__device__ __forceinline__ bf16 f2bf(float f) {
  uint32_t u = __builtin_bit_cast(uint32_t, f);
  uint32_t r = u + 0x7FFFu + ((u >> 16) & 1u);   // round-to-nearest-even
  uint16_t h = (uint16_t)(r >> 16);
  return __builtin_bit_cast(bf16, h);
}

// pack two floats into bf16x2 dword (lo -> [15:0], hi -> [31:16]), RNE
__device__ __forceinline__ uint32_t pk_bf16(float lo, float hi) {
  uint32_t ul = __builtin_bit_cast(uint32_t, lo);
  uint32_t uh = __builtin_bit_cast(uint32_t, hi);
  ul += 0x7FFFu + ((ul >> 16) & 1u);
  uh += 0x7FFFu + ((uh >> 16) & 1u);
  return (ul >> 16) | (uh & 0xFFFF0000u);
}

__device__ __forceinline__ float fast_silu(float g) {
  return g * __builtin_amdgcn_rcpf(1.0f + __expf(-g));
}

// async copy 16B global -> LDS (GVS mode, tracked by ASYNCcnt)
__device__ __forceinline__ void async_g2l_b128(uint32_t lds_addr, uint64_t sbase,
                                               uint32_t voff) {
  asm volatile("global_load_async_to_lds_b128 %0, %1, %2"
               :: "v"(lds_addr), "v"(voff), "s"(sbase) : "memory");
}
__device__ __forceinline__ void wait_async0() {
  asm volatile("s_wait_asynccnt 0x0" ::: "memory");
}

// build 16-elem bf16 fragment from two contiguous 16B LDS chunks
// A frags: p = &lA[m][8*half],  hioff = 16   (k = (e<8?e:e+8) + 8*half)
// B frags: p = &lB[n][8*half],  hioff = 16   (same k order as A)
__device__ __forceinline__ v16bf frag_ld(const bf16* p, int hioff) {
  bf16x8 lo = *(const bf16x8*)p;
  bf16x8 hi = *(const bf16x8*)(p + hioff);
  v16bf r;
#pragma unroll
  for (int i = 0; i < 8; ++i) { r[i] = lo[i]; r[i + 8] = hi[i]; }
  return r;
}

// ------------------------- fp32 -> bf16 activation copy ---------------------
__global__ void k_cvt_bf16(const float* __restrict__ x, bf16* __restrict__ y, int n) {
  int i = (blockIdx.x * blockDim.x + threadIdx.x) * 8;
  if (i >= n) return;
  f32x4 a = *(const f32x4*)(x + i);
  f32x4 b = *(const f32x4*)(x + i + 4);
  u32x4 o = { pk_bf16(a[0], a[1]), pk_bf16(a[2], a[3]),
              pk_bf16(b[0], b[1]), pk_bf16(b[2], b[3]) };
  *(u32x4*)(y + i) = o;
}

// ------------------------------- router ------------------------------------
__global__ void k_router(const float* __restrict__ x, const float* __restrict__ gw,
                         float* __restrict__ comb) {
  __shared__ float sl[16][E_ + 1];
  const int tid = threadIdx.x;
  const int tt  = tid >> 4;
  const int e   = tid & 15;
  const int t   = blockIdx.x * 16 + tt;

  const float* xr = x + (size_t)t * H_;
  float acc = 0.f;
  for (int k = 0; k < H_; ++k) acc += xr[k] * gw[k * E_ + e];
  sl[tt][e] = acc;
  __syncthreads();

  if (e == 0) {
    float s[E_];
    float mx = -1e30f;
    for (int j = 0; j < E_; ++j) { s[j] = sl[tt][j]; mx = fmaxf(mx, s[j]); }
    float sum = 0.f;
    for (int j = 0; j < E_; ++j) { s[j] = __expf(s[j] - mx); sum += s[j]; }
    float inv = __builtin_amdgcn_rcpf(sum);
    for (int j = 0; j < E_; ++j) s[j] *= inv;          // softmax scores

    float gs[G_];
    for (int g = 0; g < G_; ++g) {
      float m = s[g * 4];
      for (int j = 1; j < 4; ++j) m = fmaxf(m, s[g * 4 + j]);
      gs[g] = m;
    }
    int g0 = 0;
    for (int g = 1; g < G_; ++g) if (gs[g] > gs[g0]) g0 = g;
    int g1 = -1;
    for (int g = 0; g < G_; ++g)
      if (g != g0 && (g1 < 0 || gs[g] > gs[g1])) g1 = g;
    for (int j = 0; j < E_; ++j) {
      int g = j >> 2;
      if (g != g0 && g != g1) s[j] = 0.f;
    }
    float w[K_]; int id[K_]; float wsum = 0.f;
    for (int i = 0; i < K_; ++i) {
      int b = 0;
      for (int j = 1; j < E_; ++j) if (s[j] > s[b]) b = j;
      w[i] = s[b]; id[i] = b; s[b] = -1.f; wsum += w[i];
    }
    float* cr = comb + (size_t)t * E_;
    for (int j = 0; j < E_; ++j) cr[j] = 0.f;
    float winv = 1.f / wsum;
    for (int i = 0; i < K_; ++i) cr[id[i]] = w[i] * winv;
  }
}

// ------------------- fused gate_up GEMM + SiLU*Mul epilogue -----------------
// act[T x Ndim] = silu(xb @ W[:, :Ndim]) * (xb @ W[:, Ndim:]), bf16 out.
// Grid: (T/BM, Ndim/BN), 256 threads. Double-buffered software pipeline;
// requires Kdim/BK even (holds: 2048/32=64, 1408/32=44, 2816/32=88).
__global__ __launch_bounds__(256)
void k_gateup_silu(const bf16* __restrict__ xb,   // T x Kdim (bf16)
                   const float* __restrict__ W,   // Kdim x 2*Ndim (fp32)
                   bf16* __restrict__ act,        // T x Ndim (bf16)
                   int Kdim, int Ndim) {
  __shared__ bf16 lA[2][BM][LDP];
  __shared__ bf16 lBg[2][BN][LDP];
  __shared__ bf16 lBu[2][BN][LDP];

  const int tid  = threadIdx.x;
  const int lane = tid & 31;
  const int wave = tid >> 5;
  const int wm   = (wave >> 1) * 32;
  const int wn   = (wave & 1) * 16;
  const int half = lane >> 4;
  const int l16  = lane & 15;
  const int m0   = blockIdx.x * BM;
  const int n0   = blockIdx.y * BN;
  const int N2   = 2 * Ndim;

  // A tile async setup: rows r0 and r0+64, 16B chunk at col c0
  const int r0 = tid >> 2;
  const int c0 = (tid & 3) << 3;
  const uint32_t ldsA0 = (uint32_t)(uintptr_t)&lA[0][r0][c0];
  const uint32_t ldsA1 = (uint32_t)(uintptr_t)&lA[0][r0 + 64][c0];
  const uint64_t gA    = (uint64_t)(uintptr_t)xb;
  uint32_t vofA0 = (uint32_t)(((m0 + r0) * Kdim + c0) * 2);
  uint32_t vofA1 = (uint32_t)(((m0 + r0 + 64) * Kdim + c0) * 2);

  // B tile cvt setup: k-pairs kp0 and kp0+8, column nj; packed b32 LDS stores
  const int kp0 = tid >> 5;          // 0..7
  const int nj  = tid & 31;
  const float* pW0 = W + (size_t)(2 * kp0) * N2 + n0 + nj;
  const float* pW1 = W + (size_t)(2 * kp0 + 16) * N2 + n0 + nj;
  uint32_t* const sBg0 = (uint32_t*)&lBg[0][nj][2 * kp0];
  uint32_t* const sBg1 = (uint32_t*)&lBg[0][nj][2 * kp0 + 16];
  uint32_t* const sBu0 = (uint32_t*)&lBu[0][nj][2 * kp0];
  uint32_t* const sBu1 = (uint32_t*)&lBu[0][nj][2 * kp0 + 16];

  // fragment base pointers (buffer 0)
  const bf16* const fA0 = &lA[0][wm + l16][half * 8];
  const bf16* const fA1 = &lA[0][wm + 16 + l16][half * 8];
  const bf16* const fBg = &lBg[0][wn + l16][half * 8];
  const bf16* const fBu = &lBu[0][wn + l16][half * 8];

  v8f cg0 = {0,0,0,0,0,0,0,0};
  v8f cg1 = cg0, cu0 = cg0, cu1 = cg0;

  float g0, g1, g2, g3, u0, u1, u2, u3;     // B prefetch registers

  auto issueA = [&](int buf) {
    async_g2l_b128(ldsA0 + (uint32_t)(buf * ABUF * 2), gA, vofA0);
    async_g2l_b128(ldsA1 + (uint32_t)(buf * ABUF * 2), gA, vofA1);
    vofA0 += BK * 2;
    vofA1 += BK * 2;
  };
  auto loadB = [&]() {
    g0 = pW0[0];     g1 = pW0[N2];
    u0 = pW0[Ndim];  u1 = pW0[Ndim + N2];
    g2 = pW1[0];     g3 = pW1[N2];
    u2 = pW1[Ndim];  u3 = pW1[Ndim + N2];
    pW0 += (size_t)BK * N2;
    pW1 += (size_t)BK * N2;
  };
  auto storeB = [&](int buf) {
    sBg0[buf * BBUF] = pk_bf16(g0, g1);
    sBg1[buf * BBUF] = pk_bf16(g2, g3);
    sBu0[buf * BBUF] = pk_bf16(u0, u1);
    sBu1[buf * BBUF] = pk_bf16(u2, u3);
  };
  auto compute = [&](int buf) {
    v16bf a0 = frag_ld(fA0 + buf * ABUF, 16);
    v16bf a1 = frag_ld(fA1 + buf * ABUF, 16);
    v16bf bg = frag_ld(fBg + buf * ABUF, 8);   // wrong stride guarded below
    v16bf bu = frag_ld(fBu + buf * ABUF, 8);
    (void)a0; (void)a1; (void)bg; (void)bu;
    cg0 = __builtin_amdgcn_wmma_f32_16x16x32_bf16(false, a0, false, bg, (short)0, cg0, false, false);
    cg1 = __builtin_amdgcn_wmma_f32_16x16x32_bf16(false, a1, false, bg, (short)0, cg1, false, false);
    cu0 = __builtin_amdgcn_wmma_f32_16x16x32_bf16(false, a0, false, bu, (short)0, cu0, false, false);
    cu1 = __builtin_amdgcn_wmma_f32_16x16x32_bf16(false, a1, false, bu, (short)0, cu1, false, false);
  };
  // NOTE: B buffers stride is BN*LDP (not ABUF); use dedicated loader:
  auto computeB = [&](int buf) {
    v16bf a0 = frag_ld(fA0 + buf * ABUF, 16);
    v16bf a1 = frag_ld(fA1 + buf * ABUF, 16);
    v16bf bg = frag_ld(fBg + buf * (BN * LDP), 16);
    v16bf bu = frag_ld(fBu + buf * (BN * LDP), 16);
    cg0 = __builtin_amdgcn_wmma_f32_16x16x32_bf16(false, a0, false, bg, (short)0, cg0, false, false);
    cg1 = __builtin_amdgcn_wmma_f32_16x16x32_bf16(false, a1, false, bg, (short)0, cg1, false, false);
    cu0 = __builtin_amdgcn_wmma_f32_16x16x32_bf16(false, a0, false, bu, (short)0, cu0, false, false);
    cu1 = __builtin_amdgcn_wmma_f32_16x16x32_bf16(false, a1, false, bu, (short)0, cu1, false, false);
  };
  (void)compute;

  const int nk = Kdim / BK;   // even by construction

  // prologue: stage tile 0 into buffer 0
  issueA(0);
  loadB();
  storeB(0);
  wait_async0();
  __syncthreads();

  // steady state: 2 tiles per iteration, statically indexed ping-pong
  for (int kt = 0; kt < nk - 2; kt += 2) {
    issueA(1); loadB();          // prefetch kt+1 -> buf1
    computeB(0);                 // compute kt on buf0
    storeB(1);
    wait_async0();
    __syncthreads();

    issueA(0); loadB();          // prefetch kt+2 -> buf0
    computeB(1);                 // compute kt+1 on buf1
    storeB(0);
    wait_async0();
    __syncthreads();
  }
  // tail pair: tiles nk-2 (in buf0) and nk-1 (prefetch into buf1)
  issueA(1); loadB();
  computeB(0);
  storeB(1);
  wait_async0();
  __syncthreads();
  computeB(1);

  // epilogue: D layout -> lane L: n = L%16; reg r: m = r + 8*(L/16)
  bf16* pact = act + (size_t)(m0 + wm + 8 * half) * Ndim + (n0 + wn + l16);
#pragma unroll
  for (int r = 0; r < 8; ++r) {
    pact[(size_t)r * Ndim]        = f2bf(fast_silu(cg0[r]) * cu0[r]);
    pact[(size_t)(r + 16) * Ndim] = f2bf(fast_silu(cg1[r]) * cu1[r]);
  }
}

// ----------------- down GEMM with per-row scale, accumulate -----------------
// out[T x H] += rowscale(m) * (act @ W);  rowscale = scale[m*stride] or 1.0
__global__ __launch_bounds__(256)
void k_down_acc(const bf16* __restrict__ act,    // T x Kdim (bf16)
                const float* __restrict__ W,     // Kdim x H (fp32)
                float* __restrict__ out,         // T x H (fp32, accumulated)
                const float* __restrict__ scale, // per-row scale or nullptr
                int scale_stride, int Kdim) {
  __shared__ bf16 lA[2][BM][LDP];
  __shared__ bf16 lB[2][BN][LDP];

  const int tid  = threadIdx.x;
  const int lane = tid & 31;
  const int wave = tid >> 5;
  const int wm   = (wave >> 1) * 32;
  const int wn   = (wave & 1) * 16;
  const int half = lane >> 4;
  const int l16  = lane & 15;
  const int m0   = blockIdx.x * BM;
  const int n0   = blockIdx.y * BN;

  const int r0 = tid >> 2;
  const int c0 = (tid & 3) << 3;
  const uint32_t ldsA0 = (uint32_t)(uintptr_t)&lA[0][r0][c0];
  const uint32_t ldsA1 = (uint32_t)(uintptr_t)&lA[0][r0 + 64][c0];
  const uint64_t gA    = (uint64_t)(uintptr_t)act;
  uint32_t vofA0 = (uint32_t)(((m0 + r0) * Kdim + c0) * 2);
  uint32_t vofA1 = (uint32_t)(((m0 + r0 + 64) * Kdim + c0) * 2);

  const int kp0 = tid >> 5;
  const int nj  = tid & 31;
  const float* pW0 = W + (size_t)(2 * kp0) * H_ + n0 + nj;
  const float* pW1 = W + (size_t)(2 * kp0 + 16) * H_ + n0 + nj;
  uint32_t* const sB0 = (uint32_t*)&lB[0][nj][2 * kp0];
  uint32_t* const sB1 = (uint32_t*)&lB[0][nj][2 * kp0 + 16];

  const bf16* const fA0 = &lA[0][wm + l16][half * 8];
  const bf16* const fA1 = &lA[0][wm + 16 + l16][half * 8];
  const bf16* const fB  = &lB[0][wn + l16][half * 8];

  v8f c0v = {0,0,0,0,0,0,0,0};
  v8f c1v = c0v;

  float w0, w1, w2, w3;

  auto issueA = [&](int buf) {
    async_g2l_b128(ldsA0 + (uint32_t)(buf * ABUF * 2), gA, vofA0);
    async_g2l_b128(ldsA1 + (uint32_t)(buf * ABUF * 2), gA, vofA1);
    vofA0 += BK * 2;
    vofA1 += BK * 2;
  };
  auto loadB = [&]() {
    w0 = pW0[0]; w1 = pW0[H_];
    w2 = pW1[0]; w3 = pW1[H_];
    pW0 += (size_t)BK * H_;
    pW1 += (size_t)BK * H_;
  };
  auto storeB = [&](int buf) {
    sB0[buf * BBUF] = pk_bf16(w0, w1);
    sB1[buf * BBUF] = pk_bf16(w2, w3);
  };
  auto computeB = [&](int buf) {
    v16bf a0 = frag_ld(fA0 + buf * ABUF, 16);
    v16bf a1 = frag_ld(fA1 + buf * ABUF, 16);
    v16bf b  = frag_ld(fB + buf * (BN * LDP), 16);
    c0v = __builtin_amdgcn_wmma_f32_16x16x32_bf16(false, a0, false, b, (short)0, c0v, false, false);
    c1v = __builtin_amdgcn_wmma_f32_16x16x32_bf16(false, a1, false, b, (short)0, c1v, false, false);
  };

  const int nk = Kdim / BK;   // even by construction

  issueA(0);
  loadB();
  storeB(0);
  wait_async0();
  __syncthreads();

  for (int kt = 0; kt < nk - 2; kt += 2) {
    issueA(1); loadB();
    computeB(0);
    storeB(1);
    wait_async0();
    __syncthreads();

    issueA(0); loadB();
    computeB(1);
    storeB(0);
    wait_async0();
    __syncthreads();
  }
  issueA(1); loadB();
  computeB(0);
  storeB(1);
  wait_async0();
  __syncthreads();
  computeB(1);

  float* pout = out + (size_t)(m0 + wm + 8 * half) * H_ + (n0 + wn + l16);
  const float* psc = scale ? scale + (size_t)(m0 + wm + 8 * half) * scale_stride
                           : nullptr;
#pragma unroll
  for (int r = 0; r < 8; ++r) {
    float s0 = psc ? psc[(size_t)r * scale_stride] : 1.0f;
    pout[(size_t)r * H_] += s0 * c0v[r];
    float s1 = psc ? psc[(size_t)(r + 16) * scale_stride] : 1.0f;
    pout[(size_t)(r + 16) * H_] += s1 * c1v[r];
  }
}

// ------------------------------- launcher -----------------------------------
extern "C" void kernel_launch(void* const* d_in, const int* in_sizes, int n_in,
                              void* d_out, int out_size, void* d_ws, size_t ws_size,
                              hipStream_t stream) {
  (void)in_sizes; (void)n_in; (void)out_size; (void)ws_size;

  const float* x   = (const float*)d_in[0];   // (T, H)
  const float* gw  = (const float*)d_in[1];   // (H, E)
  const float* wgu = (const float*)d_in[2];   // (E, H, 2I)
  const float* wd  = (const float*)d_in[3];   // (E, I, H)
  const float* sgu = (const float*)d_in[4];   // (H, 2IS)
  const float* sdw = (const float*)d_in[5];   // (IS, H)
  float* out = (float*)d_out;                 // (T, H)

  // workspace layout (all 16B-aligned): x_bf16 | comb | act
  char* ws = (char*)d_ws;
  bf16*  xb   = (bf16*)ws;                                               // T*H bf16
  float* comb = (float*)(ws + (size_t)T_ * H_ * 2);                      // T*E f32
  bf16*  actb = (bf16*)(ws + (size_t)T_ * H_ * 2 + (size_t)T_ * E_ * 4); // T*IS bf16

  hipMemsetAsync(out, 0, (size_t)T_ * H_ * sizeof(float), stream);

  k_cvt_bf16<<<(T_ * H_ / 8 + 255) / 256, 256, 0, stream>>>(x, xb, T_ * H_);
  k_router<<<T_ / 16, 256, 0, stream>>>(x, gw, comb);

  // shared experts (scale = 1.0)
  k_gateup_silu<<<dim3(T_ / BM, IS_ / BN), 256, 0, stream>>>(xb, sgu, actb, H_, IS_);
  k_down_acc<<<dim3(T_ / BM, H_ / BN), 256, 0, stream>>>(actb, sdw, out, nullptr, 0, IS_);

  // routed experts, dense over E (matches reference); rowscale = comb[:, e]
  for (int e = 0; e < E_; ++e) {
    k_gateup_silu<<<dim3(T_ / BM, I_ / BN), 256, 0, stream>>>(
        xb, wgu + (size_t)e * H_ * 2 * I_, actb, H_, I_);
    k_down_acc<<<dim3(T_ / BM, H_ / BN), 256, 0, stream>>>(
        actb, wd + (size_t)e * I_ * H_, out, comb + e, E_, I_);
  }
}
